// ReducePoolings_69509750718955
// MI455X (gfx1250) — compile-verified, matrix operand order checked by reading, not run on hardware
//
#include <hip/hip_runtime.h>

#define Bn   8
#define An   16384
#define Sn   2048
#define Dn   512
#define OUTn 512

typedef _Float16 half8  __attribute__((ext_vector_type(8)));
typedef _Float16 half16 __attribute__((ext_vector_type(16)));
typedef float    floatx8 __attribute__((ext_vector_type(8)));

// ---------------------------------------------------------------------------
// 0) zero the small accumulators (avgsum[B*D], nfill[B])
// ---------------------------------------------------------------------------
__global__ void rp_init_kernel(float* __restrict__ avgsum, int* __restrict__ nfill) {
  const int t = threadIdx.x;
  for (int i = t; i < Bn * Dn; i += 256) avgsum[i] = 0.f;
  if (t < Bn) nfill[t] = 0;
}

// ---------------------------------------------------------------------------
// 1) segment-mean pooling. ids are sorted per protein, so each (b,s) block
//    binary-searches its atom range and reduces with plain coalesced loads.
//    Writes Xh (f16) and counts; accumulates nfill.
// ---------------------------------------------------------------------------
__global__ void rp_pool_kernel(const float* __restrict__ rep,
                               const int* __restrict__ ids,
                               const int* __restrict__ aa_len,
                               const int* __restrict__ seq_len,
                               _Float16* __restrict__ Xh,
                               int* __restrict__ counts,
                               int* __restrict__ nfill) {
  const int bs = blockIdx.x;
  const int b = bs / Sn, s = bs % Sn;
  const int len = aa_len[b];
  const int sl  = seq_len[b];
  const int* row = ids + b * An;

  // lower_bound(s)
  int lo = 0, hi = len;
  while (lo < hi) { int m = (lo + hi) >> 1; if (row[m] < s) lo = m + 1; else hi = m; }
  const int start = lo;
  // lower_bound(s+1)
  hi = len;
  while (lo < hi) { int m = (lo + hi) >> 1; if (row[m] <= s) lo = m + 1; else hi = m; }
  const int end = lo;
  const int cnt = end - start;

  const int t = threadIdx.x;          // 256 threads, 2 channels each
  float sum0 = 0.f, sum1 = 0.f;
  const float* base = rep + (size_t)b * An * Dn;
  for (int a = start; a < end; ++a) {
    const float* p = base + (size_t)a * Dn;
    sum0 += p[t];
    sum1 += p[t + 256];
  }
  const float inv = (cnt > 0) ? 1.f / (float)cnt : 0.f;
  _Float16* xr = Xh + (size_t)bs * Dn;
  xr[t]       = (_Float16)(sum0 * inv);
  xr[t + 256] = (_Float16)(sum1 * inv);

  if (t == 0) {
    counts[bs] = cnt;
    if (cnt > 0 && s < sl) atomicAdd(&nfill[b], 1);
  }
}

// ---------------------------------------------------------------------------
// 2a) partial sums over filled residues -> avgsum[b,d]
// ---------------------------------------------------------------------------
__global__ void rp_avgpart_kernel(const _Float16* __restrict__ Xh,
                                  const int* __restrict__ counts,
                                  const int* __restrict__ seq_len,
                                  float* __restrict__ avgsum) {
  const int b = blockIdx.x >> 3;
  const int chunk = blockIdx.x & 7;
  const int sl = seq_len[b];
  const int t = threadIdx.x;
  int s0 = chunk * (Sn / 8);
  int s1 = s0 + (Sn / 8);
  if (s1 > sl) s1 = sl;
  float sum0 = 0.f, sum1 = 0.f;
  for (int s = s0; s < s1; ++s) {
    if (counts[b * Sn + s] > 0) {
      const _Float16* xr = Xh + ((size_t)b * Sn + s) * Dn;
      sum0 += (float)xr[t];
      sum1 += (float)xr[t + 256];
    }
  }
  atomicAdd(&avgsum[b * Dn + t],       sum0);
  atomicAdd(&avgsum[b * Dn + t + 256], sum1);
}

// 2b) avg_h[b,d] = avgsum / max(nfill,1)
__global__ void rp_avgfin_kernel(const float* __restrict__ avgsum,
                                 const int* __restrict__ nfill,
                                 _Float16* __restrict__ avg_h) {
  const int i = blockIdx.x * 256 + threadIdx.x;   // B*D = 4096
  int nf = nfill[i / Dn];
  if (nf < 1) nf = 1;
  avg_h[i] = (_Float16)(avgsum[i] / (float)nf);
}

// ---------------------------------------------------------------------------
// 3) unfilled in-range rows <- avg ; out-of-range rows <- 0
// ---------------------------------------------------------------------------
__global__ void rp_fill_kernel(_Float16* __restrict__ Xh,
                               const int* __restrict__ counts,
                               const int* __restrict__ seq_len,
                               const _Float16* __restrict__ avg_h) {
  const int bs = blockIdx.x;
  const int b = bs / Sn, s = bs % Sn;
  const int sl = seq_len[b];
  const int cnt = counts[bs];
  const int t = threadIdx.x;                     // 64 threads, uint4 each (512 halves)
  uint4* xr = (uint4*)(Xh + (size_t)bs * Dn);
  if (s >= sl) {
    uint4 z; z.x = z.y = z.z = z.w = 0u;
    xr[t] = z;
  } else if (cnt == 0) {
    const uint4* ar = (const uint4*)(avg_h + b * Dn);
    xr[t] = ar[t];
  }
}

// ---------------------------------------------------------------------------
// 4) W f32 -> f16
// ---------------------------------------------------------------------------
__global__ void rp_wcvt_kernel(const float* __restrict__ W, _Float16* __restrict__ Wh) {
  const int i = (blockIdx.x * 256 + threadIdx.x) * 4;
#pragma unroll
  for (int k = 0; k < 4; ++k) Wh[i + k] = (_Float16)W[i + k];
}

// ---------------------------------------------------------------------------
// 5) WMMA GEMM: out[M=16384, N=512] = Xh[M,K=512] * Wh[N,K]^T + bias
//    Block = 8 waves covering 128x128; each wave: 4 M-tiles x 2 N-tiles.
//    A/B fragment layout per CDNA5 ISA (16-bit, 16x32):
//      lanes 0-15 : K = {0..7} then {16..23}
//      lanes16-31 : K = {8..15} then {24..31}
// ---------------------------------------------------------------------------
__global__ void __launch_bounds__(256, 1)
rp_gemm_kernel(const _Float16* __restrict__ Xh,
               const _Float16* __restrict__ Wh,
               const float* __restrict__ bias,
               float* __restrict__ out) {
  const int lane = threadIdx.x & 31;
  const int w    = threadIdx.x >> 5;
  const int lm   = lane & 15;
  const int lsel = (lane < 16) ? 0 : 8;

  const int M0 = blockIdx.x * 128 + (w & 1) * 64;   // wave's 64-row strip
  const int N0 = blockIdx.y * 128 + (w >> 1) * 32;  // wave's 32-col strip

  floatx8 acc[4][2];
  const floatx8 zero = {0.f, 0.f, 0.f, 0.f, 0.f, 0.f, 0.f, 0.f};
#pragma unroll
  for (int i = 0; i < 4; ++i)
#pragma unroll
    for (int j = 0; j < 2; ++j) acc[i][j] = zero;

  const _Float16* arow[4];
#pragma unroll
  for (int i = 0; i < 4; ++i)
    arow[i] = Xh + (size_t)(M0 + i * 16 + lm) * Dn + lsel;
  const _Float16* brow[2];
#pragma unroll
  for (int j = 0; j < 2; ++j)
    brow[j] = Wh + (size_t)(N0 + j * 16 + lm) * Dn + lsel;

  for (int k0 = 0; k0 < Dn; k0 += 32) {
    half16 afr[4], bfr[2];
#pragma unroll
    for (int i = 0; i < 4; ++i) {
      half8 alo = *(const half8*)(arow[i] + k0);
      half8 ahi = *(const half8*)(arow[i] + k0 + 16);
      afr[i] = __builtin_shufflevector(alo, ahi,
               0, 1, 2, 3, 4, 5, 6, 7, 8, 9, 10, 11, 12, 13, 14, 15);
    }
#pragma unroll
    for (int j = 0; j < 2; ++j) {
      half8 blo = *(const half8*)(brow[j] + k0);
      half8 bhi = *(const half8*)(brow[j] + k0 + 16);
      bfr[j] = __builtin_shufflevector(blo, bhi,
               0, 1, 2, 3, 4, 5, 6, 7, 8, 9, 10, 11, 12, 13, 14, 15);
    }
#pragma unroll
    for (int i = 0; i < 4; ++i)
#pragma unroll
      for (int j = 0; j < 2; ++j)
        acc[i][j] = __builtin_amdgcn_wmma_f32_16x16x32_f16(
            false, afr[i], false, bfr[j], (short)0, acc[i][j], false, false);
  }

  // epilogue: C layout => VGPR v: lanes0-15 row=v, lanes16-31 row=v+8; col=lane&15
#pragma unroll
  for (int j = 0; j < 2; ++j) {
    const int n = N0 + j * 16 + lm;
    const float bv = bias[n];
#pragma unroll
    for (int i = 0; i < 4; ++i) {
#pragma unroll
      for (int v = 0; v < 8; ++v) {
        const int m = M0 + i * 16 + ((lane < 16) ? v : v + 8);
        out[(size_t)m * OUTn + n] = acc[i][j][v] + bv;
      }
    }
  }
}

// ---------------------------------------------------------------------------
extern "C" void kernel_launch(void* const* d_in, const int* in_sizes, int n_in,
                              void* d_out, int out_size, void* d_ws, size_t ws_size,
                              hipStream_t stream) {
  (void)in_sizes; (void)n_in; (void)out_size; (void)ws_size;
  const float* rep    = (const float*)d_in[0];
  const int*   ids    = (const int*)d_in[1];
  const int*   aalen  = (const int*)d_in[2];
  const int*   seqlen = (const int*)d_in[3];
  const float* W      = (const float*)d_in[4];
  const float* bias   = (const float*)d_in[5];
  float* out = (float*)d_out;

  char* ws = (char*)d_ws;
  _Float16* Xh     = (_Float16*)(ws);                         // 16 MiB
  _Float16* Wh     = (_Float16*)(ws + 16777216);              // 512 KiB
  int*      counts = (int*)     (ws + 16777216 + 524288);     // 64 KiB
  float*    avgsum = (float*)   (ws + 17367040);              // 16 KiB
  _Float16* avg_h  = (_Float16*)(ws + 17383424);              // 8 KiB
  int*      nfill  = (int*)     (ws + 17391616);              // 32 B

  rp_init_kernel<<<1, 256, 0, stream>>>(avgsum, nfill);
  rp_pool_kernel<<<Bn * Sn, 256, 0, stream>>>(rep, ids, aalen, seqlen, Xh, counts, nfill);
  rp_avgpart_kernel<<<Bn * 8, 256, 0, stream>>>(Xh, counts, seqlen, avgsum);
  rp_avgfin_kernel<<<(Bn * Dn) / 256, 256, 0, stream>>>(avgsum, nfill, avg_h);
  rp_fill_kernel<<<Bn * Sn, 64, 0, stream>>>(Xh, counts, seqlen, avg_h);
  rp_wcvt_kernel<<<(OUTn * Dn) / 1024, 256, 0, stream>>>(W, Wh);
  rp_gemm_kernel<<<dim3((Bn * Sn) / 128, OUTn / 128), 256, 0, stream>>>(Xh, Wh, bias, out);
}
